// Hgru3_19550691131819
// MI455X (gfx1250) — matrix operand order repaired and checked by reading, not compiled, not run on hardware
//
#include <hip/hip_runtime.h>

// ---------------------------------------------------------------------------
// Types / helpers
// ---------------------------------------------------------------------------
typedef __attribute__((ext_vector_type(16))) __bf16        bf16x16;
typedef __attribute__((ext_vector_type(8)))  float         f32x8;
typedef __attribute__((ext_vector_type(4)))  unsigned int  u32x4;
typedef __attribute__((ext_vector_type(8)))  int           i32x8;
typedef __attribute__((ext_vector_type(4)))  int           i32x4;

union FragU {
  bf16x16 bf;
  u32x4   q[2];
  unsigned short us[16];
};

__device__ __forceinline__ unsigned short f2bf(float f) {
  __bf16 h = (__bf16)f;                       // native v_cvt to bf16
  return __builtin_bit_cast(unsigned short, h);
}
__device__ __forceinline__ float bf2f(unsigned short s) {
  return __uint_as_float(((unsigned int)s) << 16);
}

// A fragment: 16x32 (MxK) bf16 from row-major [.,lda] tile at (m0,k0).
__device__ __forceinline__ bf16x16 load_frag_a(const unsigned short* base, int lda,
                                               int m0, int k0, int lane) {
  int m = m0 + (lane & 15);
  int h = lane >> 4;
  const unsigned short* p = base + (size_t)m * lda + k0 + h * 8;
  FragU f;
  f.q[0] = *(const u32x4*)p;
  f.q[1] = *(const u32x4*)(p + 16);
  return f.bf;
}

// B fragment: 32x16 (KxN) bf16 sourced from Bt[n][k] row-major [.,ldb] at (n0,k0).
__device__ __forceinline__ bf16x16 load_frag_b(const unsigned short* base, int ldb,
                                               int n0, int k0, int lane) {
  int n = n0 + (lane & 15);
  int h = lane >> 4;
  const unsigned short* p = base + (size_t)n * ldb + k0 + h * 8;
  FragU f;
  f.q[0] = *(const u32x4*)p;
  f.q[1] = *(const u32x4*)(p + 16);
  return f.bf;
}

__device__ __forceinline__ f32x8 wmma_bf16(bf16x16 a, bf16x16 b, f32x8 c) {
  return __builtin_amdgcn_wmma_f32_16x16x32_bf16(false, a, false, b, (short)0, c,
                                                 false, false);
}

// ---------------------------------------------------------------------------
// Tensor Data Mover: async 2D bf16 tile load global -> LDS (with LDS padding).
// Descriptor per CDNA5 ISA ch.8: group0 = {flags, lds_addr, gaddr_lo, gaddr_hi|type},
// group1 = {cfg, tensor dims/tile dims/stride}. Issued by one wave, tracked by
// TENSORcnt. This toolchain exposes the 6-arg builtin (extra int32x8 group).
// Fallback: synchronous wave-copy if builtin unavailable.
// ---------------------------------------------------------------------------
#if __has_builtin(__builtin_amdgcn_tensor_load_to_lds)
#define HGRU_TDM 1
#else
#define HGRU_TDM 0
#endif

__device__ __forceinline__ void tdm_load_2d(const unsigned short* g, unsigned short* lds,
                                            unsigned tile_w, unsigned tile_h,
                                            unsigned stride, unsigned lds_stride,
                                            unsigned padi, unsigned pada, int lane) {
#if HGRU_TDM
  (void)lane; (void)lds_stride;
  unsigned long long ga = (unsigned long long)(uintptr_t)g;
  unsigned ldsa = (unsigned)(uintptr_t)lds;          // low 32 bits = LDS byte offset
  u32x4 d0;
  d0[0] = 1u;                                        // count=1, user descriptor
  d0[1] = ldsa;                                      // lds_addr [63:32]
  d0[2] = (unsigned)ga;                              // global_addr lo
  d0[3] = (unsigned)(ga >> 32) | 0x80000000u;        // global_addr hi | type=2
  i32x8 d1;
  d1[0] = (int)((1u << 16) | (1u << 20) | (padi << 22) | (pada << 25)); // 2B elems, pad
  d1[1] = (int)((stride & 0xFFFFu) << 16);           // tensor_dim0 lo16
  d1[2] = (int)((stride >> 16) | ((1048576u & 0xFFFFu) << 16));  // dim0 hi | dim1 lo16
  d1[3] = (int)((1048576u >> 16) | (tile_w << 16));  // dim1 hi16 | tile_dim0
  d1[4] = (int)tile_h;                               // tile_dim1 (tile_dim2 = 0)
  d1[5] = (int)stride;                               // tensor_dim0_stride lo32
  d1[6] = 0;
  d1[7] = 0;
  i32x4 z4 = {0, 0, 0, 0};
  i32x8 z8 = {0, 0, 0, 0, 0, 0, 0, 0};
  __builtin_amdgcn_tensor_load_to_lds(d0, d1, z4, z4, z8, 0);
#else
  unsigned wq = tile_w >> 3;
  for (unsigned i = lane; i < wq * tile_h; i += 32) {
    unsigned r = i / wq, c = (i - r * wq) * 8;
    *(u32x4*)&lds[(size_t)r * lds_stride + c] = *(const u32x4*)&g[(size_t)r * stride + c];
  }
#endif
}

__device__ __forceinline__ void tdm_wait2() {
#if __has_builtin(__builtin_amdgcn_s_wait_tensorcnt)
  __builtin_amdgcn_s_wait_tensorcnt(2);
#endif
}
__device__ __forceinline__ void tdm_wait0() {
#if __has_builtin(__builtin_amdgcn_s_wait_tensorcnt)
  __builtin_amdgcn_s_wait_tensorcnt(0);
#endif
}

// ---------------------------------------------------------------------------
// Prep kernels
// ---------------------------------------------------------------------------
__global__ void convert_bf16_kernel(const float* __restrict__ X,
                                    unsigned short* __restrict__ Y, size_t n) {
  size_t t = (size_t)blockIdx.x * 256 + threadIdx.x;
  if (t < n) Y[t] = f2bf(X[t]);
}

// W [K][N] fp32 row-major  ->  Wt [N][K] bf16 row-major
__global__ void transpose_bf16_kernel(const float* __restrict__ W,
                                      unsigned short* __restrict__ Wt, int K, int N) {
  int t = blockIdx.x * 256 + threadIdx.x;
  if (t >= K * N) return;
  int k = t / N, n = t - k * N;
  Wt[(size_t)n * K + k] = f2bf(W[(size_t)k * N + n]);
}

// ---------------------------------------------------------------------------
// bf16 WMMA GEMM: C[M,N] = act(A[M,K] * B[K,N]), Bt given as [N][K].
// 256 thr / 8 waves, tile 128x128, BK=32, TDM double-buffered staging.
// Wave (rw=w>>1, cw=w&1) owns 32x64: 2 A-frags x 4 B-frags -> 8 WMMA / K-step.
// ---------------------------------------------------------------------------
template <int ACT, bool OBF>
__global__ void __launch_bounds__(256)
gemm_bf16(const unsigned short* __restrict__ A, const unsigned short* __restrict__ Bt,
          void* __restrict__ Cv, int M, int N, int K) {
  __shared__ unsigned short sA[2][128][40];
  __shared__ unsigned short sB[2][128][40];
  int tid = threadIdx.x, lane = tid & 31, w = tid >> 5;
  int rw = w >> 1, cw = w & 1;
  int m0 = blockIdx.y * 128, n0 = blockIdx.x * 128;
  int nt = K >> 5;

  if (tid < 32) {
    tdm_load_2d(A + (size_t)m0 * K, &sA[0][0][0], 32, 128, K, 40, 3, 3, lane);
    tdm_load_2d(Bt + (size_t)n0 * K, &sB[0][0][0], 32, 128, K, 40, 3, 3, lane);
  }
  f32x8 acc[2][4] = {};
  for (int it = 0; it < nt; ++it) {
    int cur = it & 1;
    if (tid < 32) {
      if (it + 1 < nt) {
        int k0n = (it + 1) << 5;
        tdm_load_2d(A + (size_t)m0 * K + k0n, &sA[cur ^ 1][0][0], 32, 128, K, 40, 3, 3, lane);
        tdm_load_2d(Bt + (size_t)n0 * K + k0n, &sB[cur ^ 1][0][0], 32, 128, K, 40, 3, 3, lane);
        tdm_wait2();          // in-order: <=2 outstanding => current tile landed
      } else {
        tdm_wait0();
      }
    }
    __syncthreads();
    bf16x16 af0 = load_frag_a(&sA[cur][0][0], 40, 32 * rw, 0, lane);
    bf16x16 af1 = load_frag_a(&sA[cur][0][0], 40, 32 * rw + 16, 0, lane);
#pragma unroll
    for (int j = 0; j < 4; ++j) {
      bf16x16 bfr = load_frag_b(&sB[cur][0][0], 40, 64 * cw + 16 * j, 0, lane);
      acc[0][j] = wmma_bf16(af0, bfr, acc[0][j]);
      acc[1][j] = wmma_bf16(af1, bfr, acc[1][j]);
    }
    __syncthreads();          // buffer free before wave0 re-issues into it
  }
#pragma unroll
  for (int i = 0; i < 2; ++i) {
    int mrow = m0 + 32 * rw + 16 * i + (lane >> 4) * 8;
#pragma unroll
    for (int j = 0; j < 4; ++j) {
      int nc = n0 + 64 * cw + 16 * j + (lane & 15);
#pragma unroll
      for (int v = 0; v < 8; ++v) {
        float x = acc[i][j][v];
        if (ACT == 1) x = x / (1.f + __expf(-x));       // silu
        if (ACT == 2) x = 1.f / (1.f + __expf(-x));     // sigmoid
        size_t idx = (size_t)(mrow + v) * N + nc;
        if (OBF) ((unsigned short*)Cv)[idx] = f2bf(x);
        else     ((float*)Cv)[idx] = x;
      }
    }
  }
}

// ---------------------------------------------------------------------------
// f = x @ Wf ; log_f = log(lb + (1-lb)*sigmoid(f)), layout [B,H,N]
// ---------------------------------------------------------------------------
__global__ void __launch_bounds__(256)
fdecay_kernel(const float* __restrict__ x, const float* __restrict__ Wf,
              const float* __restrict__ llb, float* __restrict__ lf) {
  int t = blockIdx.x * 256 + threadIdx.x;           // [0, 16384*8)
  int row = t >> 3, h = t & 7;
  const float* xr = x + (size_t)row * 1024;
  float acc = 0.f;
  for (int k = 0; k < 1024; ++k) acc += xr[k] * Wf[k * 8 + h];
  float lb = __expf(llb[h]);
  float fd = lb + (1.f - lb) / (1.f + __expf(-acc));
  int b = row >> 12, n = row & 4095;
  lf[((size_t)(b * 8 + h)) * 4096 + n] = __logf(fd);
}

// ---------------------------------------------------------------------------
// Householder: q' = SCALE * (q - 2*(q.beta)/max(|beta|,1e-6)^2 * beta) -> bf16
// ---------------------------------------------------------------------------
__global__ void __launch_bounds__(256)
householder_kernel(const float* __restrict__ q, const float* __restrict__ beta,
                   unsigned short* __restrict__ qout) {
  __shared__ float r0[256], r1[256];
  size_t r = blockIdx.x;
  const float* qr = q + r * 1024;
  const float* br = beta + r * 1024;
  float d = 0.f, nb = 0.f;
  for (int i = threadIdx.x; i < 1024; i += 256) {
    float qq = qr[i], bb = br[i];
    d += qq * bb;
    nb += bb * bb;
  }
  r0[threadIdx.x] = d; r1[threadIdx.x] = nb;
  __syncthreads();
  for (int off = 128; off; off >>= 1) {
    if (threadIdx.x < off) { r0[threadIdx.x] += r0[threadIdx.x + off];
                             r1[threadIdx.x] += r1[threadIdx.x + off]; }
    __syncthreads();
  }
  float nrm = fmaxf(sqrtf(r1[0]), 1e-6f);
  float c = 2.f * r0[0] / (nrm * nrm);
  const float scale = 0.08838834764831845f;   // 128^-0.5
  for (int i = threadIdx.x; i < 1024; i += 256)
    qout[r * 1024 + i] = f2bf((qr[i] - c * br[i]) * scale);
}

// ---------------------------------------------------------------------------
// Chunked gated linear attention. Grid = B*H*2 (dv split in halves of 64).
// q/k chunk tiles arrive via TDM (padded stride 136); state S [128x64] lives in
// per-wave WMMA accumulators; S^T mirrored in LDS as bf16 for the q.S matmul.
// ---------------------------------------------------------------------------
__global__ void __launch_bounds__(256)
attn_kernel(const unsigned short* __restrict__ qf, const unsigned short* __restrict__ kf,
            const unsigned short* __restrict__ vf, const float* __restrict__ lf,
            const float* __restrict__ gate, unsigned short* __restrict__ og) {
  extern __shared__ __align__(16) unsigned char smem[];
  unsigned short* sQ  = (unsigned short*)smem;   // [64][136]  q (TDM)
  unsigned short* sQg = sQ  + 64 * 136;          // [64][136]  q * exp(gcs_i)
  unsigned short* sK  = sQg + 64 * 136;          // [64][136]  k (TDM)
  unsigned short* sKt = sK  + 64 * 136;          // [128][72]  (k * exp(gtot-gcs))^T
  unsigned short* sVt = sKt + 128 * 72;          // [64][72]   v^T (dv slice)
  unsigned short* sAm = sVt + 64 * 72;           // [64][72]   masked A (bf16)
  unsigned short* sSt = sAm + 64 * 72;           // [64][136]  S^T: [dv][dk]
  float* sG = (float*)(sSt + 64 * 136);          // [64] inclusive cumsum of log decay

  int tid = threadIdx.x, lane = tid & 31, w = tid >> 5;
  int s  = blockIdx.x & 1;
  int bh = blockIdx.x >> 1;
  int b = bh >> 3, h = bh & 7;
  const size_t rowbase = (size_t)b * 4096;
  const int colq = h * 128;
  const int colv = h * 128 + s * 64;

  for (int i = tid; i < 64 * 136 / 2; i += 256) ((unsigned int*)sSt)[i] = 0u;
  f32x8 Sacc[4] = {};    // wave w: dk rows [16w,16w+16) x dv tiles 0..3

  for (int ch = 0; ch < 64; ++ch) {
    int t0 = ch * 64;
    __syncthreads();
    // ---- kick off TDM for q/k chunk tiles (64x128, row stride 1024) ----
    if (tid < 32) {
      tdm_load_2d(qf + (rowbase + t0) * 1024 + colq, sQ, 128, 64, 1024, 136, 5, 3, lane);
      tdm_load_2d(kf + (rowbase + t0) * 1024 + colq, sK, 128, 64, 1024, 136, 5, 3, lane);
    }
    // ---- log-decay chunk + inclusive scan (overlaps with TDM) ----
    if (tid < 64) sG[tid] = lf[(size_t)bh * 4096 + t0 + tid];
    for (int off = 1; off < 64; off <<= 1) {
      __syncthreads();
      float vsh = 0.f;
      if (tid < 64 && tid >= off) vsh = sG[tid - off];
      __syncthreads();
      if (tid < 64) sG[tid] += vsh;
    }
    if (tid < 32) tdm_wait0();
    __syncthreads();
    float gtot = sG[63];

    // ---- derive qg, kdec^T from LDS; stage v^T from global ----
    for (int slot = tid; slot < 1024; slot += 256) {
      int r = slot >> 4, c = (slot & 15) * 8;
      float eg  = __expf(sG[r]);
      float egk = __expf(gtot - sG[r]);
      u32x4 qv = *(const u32x4*)&sQ[r * 136 + c];
      u32x4 kv = *(const u32x4*)&sK[r * 136 + c];
      const unsigned short* qs = (const unsigned short*)&qv;
      const unsigned short* ks = (const unsigned short*)&kv;
      unsigned short qg[8];
#pragma unroll
      for (int i = 0; i < 8; ++i) {
        qg[i] = f2bf(bf2f(qs[i]) * eg);
        sKt[(c + i) * 72 + r] = f2bf(bf2f(ks[i]) * egk);
      }
      *(u32x4*)&sQg[r * 136 + c] = *(u32x4*)qg;
      if (slot < 512) {
        int rv = slot >> 3, cv = (slot & 7) * 8;
        u32x4 vv = *(const u32x4*)&vf[(rowbase + t0 + rv) * 1024 + colv + cv];
        const unsigned short* vs = (const unsigned short*)&vv;
#pragma unroll
        for (int i = 0; i < 8; ++i) sVt[(cv + i) * 72 + rv] = vs[i];
      }
    }
    __syncthreads();

    // ---- A = (q k^T) .* tril(exp(gcs_i - gcs_j)) -> sAm (bf16) ----
    {
      int rw = w >> 1, cw = w & 1;
      int m0 = rw * 16;
#pragma unroll
      for (int j = 0; j < 2; ++j) {
        int n0 = cw * 32 + j * 16;
        f32x8 a = {};
#pragma unroll
        for (int kk = 0; kk < 128; kk += 32) {
          bf16x16 af  = load_frag_a(sQ, 136, m0, kk, lane);
          bf16x16 bfr = load_frag_b(sK, 136, n0, kk, lane);
          a = wmma_bf16(af, bfr, a);
        }
        int mm = m0 + (lane >> 4) * 8;
        int nn = n0 + (lane & 15);
        float gn = sG[nn];
#pragma unroll
        for (int v = 0; v < 8; ++v) {
          int mi = mm + v;
          float val = (nn <= mi) ? a[v] * __expf(sG[mi] - gn) : 0.f;
          sAm[mi * 72 + nn] = f2bf(val);
        }
      }
    }
    __syncthreads();

    // ---- o = A.v + (q e^gcs).S_prev ; apply gate; store bf16 ----
    {
      int rw = w >> 1, cw = w & 1;
      int m0 = rw * 16;
#pragma unroll
      for (int j = 0; j < 2; ++j) {
        int n0 = (cw * 2 + j) * 16;
        f32x8 a = {};
#pragma unroll
        for (int kk = 0; kk < 64; kk += 32) {
          bf16x16 af  = load_frag_a(sAm, 72, m0, kk, lane);
          bf16x16 bfr = load_frag_b(sVt, 72, n0, kk, lane);
          a = wmma_bf16(af, bfr, a);
        }
#pragma unroll
        for (int kk = 0; kk < 128; kk += 32) {
          bf16x16 af  = load_frag_a(sQg, 136, m0, kk, lane);
          bf16x16 bfr = load_frag_b(sSt, 136, n0, kk, lane);
          a = wmma_bf16(af, bfr, a);
        }
        int mm = m0 + (lane >> 4) * 8;
        int nn = n0 + (lane & 15);
#pragma unroll
        for (int v = 0; v < 8; ++v) {
          size_t gi = (rowbase + t0 + mm + v) * 1024 + colv + nn;
          og[gi] = f2bf(a[v] * gate[gi]);
        }
      }
    }
    __syncthreads();

    // ---- S = e^gtot * S + (k e^{gtot-gcs})^T . v ; refresh sSt ----
    {
      float egt = __expf(gtot);
      int m0 = w * 16;                 // dk strip of this wave
#pragma unroll
      for (int j = 0; j < 4; ++j) {
#pragma unroll
        for (int v = 0; v < 8; ++v) Sacc[j][v] *= egt;
#pragma unroll
        for (int kk = 0; kk < 64; kk += 32) {
          bf16x16 af  = load_frag_a(sKt, 72, m0, kk, lane);
          bf16x16 bfr = load_frag_b(sVt, 72, j * 16, kk, lane);
          Sacc[j] = wmma_bf16(af, bfr, Sacc[j]);
        }
      }
#pragma unroll
      for (int j = 0; j < 4; ++j) {
        int nn = j * 16 + (lane & 15);
        int mk = m0 + (lane >> 4) * 8;
        unsigned short tmp[8];
#pragma unroll
        for (int v = 0; v < 8; ++v) tmp[v] = f2bf(Sacc[j][v]);
        *(u32x4*)&sSt[nn * 136 + mk] = *(u32x4*)tmp;
      }
    }
  }
}

// ---------------------------------------------------------------------------
// LayerNorm over last dim (1024)
// ---------------------------------------------------------------------------
__global__ void __launch_bounds__(256)
layernorm_kernel(const float* __restrict__ y, const float* __restrict__ gamma,
                 const float* __restrict__ betap, float* __restrict__ out) {
  __shared__ float red[256];
  size_t r = blockIdx.x;
  const float* yr = y + r * 1024;
  float sum = 0.f;
  for (int i = threadIdx.x; i < 1024; i += 256) sum += yr[i];
  red[threadIdx.x] = sum; __syncthreads();
  for (int off = 128; off; off >>= 1) {
    if (threadIdx.x < off) red[threadIdx.x] += red[threadIdx.x + off];
    __syncthreads();
  }
  float mu = red[0] * (1.f / 1024.f);
  __syncthreads();
  float v2 = 0.f;
  for (int i = threadIdx.x; i < 1024; i += 256) { float d = yr[i] - mu; v2 += d * d; }
  red[threadIdx.x] = v2; __syncthreads();
  for (int off = 128; off; off >>= 1) {
    if (threadIdx.x < off) red[threadIdx.x] += red[threadIdx.x + off];
    __syncthreads();
  }
  float rstd = rsqrtf(red[0] * (1.f / 1024.f) + 1e-5f);
  for (int i = threadIdx.x; i < 1024; i += 256)
    out[r * 1024 + i] = (yr[i] - mu) * rstd * gamma[i] + betap[i];
}

// ---------------------------------------------------------------------------
// Host launch
// ---------------------------------------------------------------------------
extern "C" void kernel_launch(void* const* d_in, const int* in_sizes, int n_in,
                              void* d_out, int out_size, void* d_ws, size_t ws_size,
                              hipStream_t stream) {
  (void)in_sizes; (void)n_in; (void)out_size; (void)ws_size;
  const float* x    = (const float*)d_in[0];
  const float* llb  = (const float*)d_in[1];
  const float* Wq   = (const float*)d_in[2];
  const float* Wk   = (const float*)d_in[3];
  const float* Wv   = (const float*)d_in[4];
  const float* Wf   = (const float*)d_in[5];
  const float* Wb1  = (const float*)d_in[6];
  const float* Wb2  = (const float*)d_in[7];
  const float* Wg1  = (const float*)d_in[8];
  const float* Wg2  = (const float*)d_in[9];
  const float* Wout = (const float*)d_in[10];
  const float* lng  = (const float*)d_in[11];
  const float* lnb  = (const float*)d_in[12];
  float* out = (float*)d_out;

  const int M = 16384, E = 1024, Rr = 128;
  char* ws = (char*)d_ws;
  size_t off = 0;
  auto alloc = [&](size_t bytes) -> char* {
    char* p = ws + off;
    off = (off + bytes + 255) & ~(size_t)255;
    return p;
  };
  unsigned short* xb    = (unsigned short*)alloc((size_t)M * E * 2);
  unsigned short* wqT   = (unsigned short*)alloc((size_t)E * E * 2);
  unsigned short* wkT   = (unsigned short*)alloc((size_t)E * E * 2);
  unsigned short* wvT   = (unsigned short*)alloc((size_t)E * E * 2);
  unsigned short* woT   = (unsigned short*)alloc((size_t)E * E * 2);
  unsigned short* wb1T  = (unsigned short*)alloc((size_t)Rr * E * 2);
  unsigned short* wb2T  = (unsigned short*)alloc((size_t)E * Rr * 2);
  unsigned short* wg1T  = (unsigned short*)alloc((size_t)Rr * E * 2);
  unsigned short* wg2T  = (unsigned short*)alloc((size_t)E * Rr * 2);
  float*          qs    = (float*)alloc((size_t)M * E * 4);
  unsigned short* kfb   = (unsigned short*)alloc((size_t)M * E * 2);
  unsigned short* vfb   = (unsigned short*)alloc((size_t)M * E * 2);
  unsigned short* b1    = (unsigned short*)alloc((size_t)M * Rr * 2);
  float*          beta  = (float*)alloc((size_t)M * E * 4);
  unsigned short* g1    = (unsigned short*)alloc((size_t)M * Rr * 2);
  float*          gateb = (float*)alloc((size_t)M * E * 4);
  unsigned short* qfb   = (unsigned short*)alloc((size_t)M * E * 2);
  float*          lfb   = (float*)alloc((size_t)M * 8 * 4);       // [B,H,N]
  unsigned short* ogb   = (unsigned short*)alloc((size_t)M * E * 2);
  float*          ybuf  = (float*)alloc((size_t)M * E * 4);

  dim3 blk(256);
  // prep
  convert_bf16_kernel<<<(unsigned)(((size_t)M * E + 255) / 256), blk, 0, stream>>>(x, xb, (size_t)M * E);
  transpose_bf16_kernel<<<(E * E + 255) / 256, blk, 0, stream>>>(Wq, wqT, E, E);
  transpose_bf16_kernel<<<(E * E + 255) / 256, blk, 0, stream>>>(Wk, wkT, E, E);
  transpose_bf16_kernel<<<(E * E + 255) / 256, blk, 0, stream>>>(Wv, wvT, E, E);
  transpose_bf16_kernel<<<(E * E + 255) / 256, blk, 0, stream>>>(Wout, woT, E, E);
  transpose_bf16_kernel<<<(E * Rr + 255) / 256, blk, 0, stream>>>(Wb1, wb1T, E, Rr);
  transpose_bf16_kernel<<<(Rr * E + 255) / 256, blk, 0, stream>>>(Wb2, wb2T, Rr, E);
  transpose_bf16_kernel<<<(E * Rr + 255) / 256, blk, 0, stream>>>(Wg1, wg1T, E, Rr);
  transpose_bf16_kernel<<<(Rr * E + 255) / 256, blk, 0, stream>>>(Wg2, wg2T, Rr, E);

  // projections
  gemm_bf16<1, false><<<dim3(E / 128, M / 128), blk, 0, stream>>>(xb, wqT, qs, M, E, E);
  gemm_bf16<1, true ><<<dim3(E / 128, M / 128), blk, 0, stream>>>(xb, wkT, kfb, M, E, E);
  gemm_bf16<0, true ><<<dim3(E / 128, M / 128), blk, 0, stream>>>(xb, wvT, vfb, M, E, E);
  gemm_bf16<0, true ><<<dim3(Rr / 128, M / 128), blk, 0, stream>>>(xb, wb1T, b1, M, Rr, E);
  gemm_bf16<1, false><<<dim3(E / 128, M / 128), blk, 0, stream>>>(b1, wb2T, beta, M, E, Rr);
  gemm_bf16<0, true ><<<dim3(Rr / 128, M / 128), blk, 0, stream>>>(xb, wg1T, g1, M, Rr, E);
  gemm_bf16<2, false><<<dim3(E / 128, M / 128), blk, 0, stream>>>(g1, wg2T, gateb, M, E, Rr);

  // decay + householder
  fdecay_kernel<<<(M * 8) / 256, blk, 0, stream>>>(x, Wf, llb, lfb);
  householder_kernel<<<M, blk, 0, stream>>>(qs, beta, qfb);

  // attention (dynamic LDS: 53248 halves + 64 floats = 106752 B)
  const unsigned attn_smem = (64 * 136 * 3 + 128 * 72 + 64 * 72 * 2 + 64 * 136) * 2 + 64 * 4;
  attn_kernel<<<64, blk, attn_smem, stream>>>(qfb, kfb, vfb, lfb, gateb, ogb);

  // output projection + layernorm
  gemm_bf16<0, false><<<dim3(E / 128, M / 128), blk, 0, stream>>>(ogb, woT, ybuf, M, E, E);
  layernorm_kernel<<<M, blk, 0, stream>>>(ybuf, lng, lnb, out);
}
